// GatedDeltaNetVector_4887672783153
// MI455X (gfx1250) — matrix operand order repaired and checked by reading, not compiled, not run on hardware
//
#include <hip/hip_runtime.h>

typedef __attribute__((ext_vector_type(16))) __bf16 v16bf;
typedef __attribute__((ext_vector_type(8)))  __bf16 v8bf;
typedef __attribute__((ext_vector_type(4)))  __bf16 v4bf;
typedef __attribute__((ext_vector_type(8)))  float  v8f;

#define B_  4
#define T_  4096
#define D_  1024
#define BT_ (B_ * T_)
#define LS  40   // padded LDS row stride (elements) for a 32-element K-slice row

#if __has_builtin(__builtin_amdgcn_global_load_async_to_lds_b128)
#define ASYNC_LDS 1
#else
#define ASYNC_LDS 0
#endif

// GCC-style int4 vector, matching the async-load builtin's pointee type exactly
typedef int v4i_t __attribute__((vector_size(16)));
typedef __attribute__((address_space(1))) v4i_t* gv4i_ptr;  // global (device) int4*
typedef __attribute__((address_space(3))) v4i_t* lv4i_ptr;  // LDS int4*

// async 16B/lane global -> LDS copy (tracked by ASYNCcnt), with sync fallback
__device__ __forceinline__ void ag2l128(const __bf16* g, __bf16* l) {
#if ASYNC_LDS
  __builtin_amdgcn_global_load_async_to_lds_b128((gv4i_ptr)g, (lv4i_ptr)l, 0, 0);
#else
  *(v8bf*)l = *(const v8bf*)g;
#endif
}
__device__ __forceinline__ void wait_async() {
#if ASYNC_LDS
#if __has_builtin(__builtin_amdgcn_s_wait_asynccnt)
  __builtin_amdgcn_s_wait_asynccnt(0);
#else
  asm volatile("s_wait_asynccnt 0" ::: "memory");
#endif
#endif
}

// ---------------- helpers ----------------

__device__ __forceinline__ __bf16 f2bf(float f) {
  unsigned u = __builtin_bit_cast(unsigned, f);
  unsigned r = u + 0x7FFFu + ((u >> 16) & 1u);   // round-to-nearest-even
  unsigned short h = (unsigned short)(r >> 16);
  return __builtin_bit_cast(__bf16, h);
}
__device__ __forceinline__ float bfu2f(unsigned short h) {
  unsigned u = ((unsigned)h) << 16;
  return __builtin_bit_cast(float, u);
}
__device__ __forceinline__ float sigm(float x) { return 1.0f / (1.0f + __expf(-x)); }
__device__ __forceinline__ float tanh_fast(float x) { return 2.0f * sigm(2.0f * x) - 1.0f; }

// A-fragment (16x32, M x K): lane l holds row (l&15); K chunks at (l>>4)*8 and (l>>4)*8+16
__device__ __forceinline__ v16bf fragA(const __bf16* sm, int lane) {
  int row = lane & 15;
  int kb  = (lane >> 4) << 3;
  const __bf16* p = sm + row * LS + kb;
  v8bf lo = *(const v8bf*)p;
  v8bf hi = *(const v8bf*)(p + 16);
  v16bf r;
#pragma unroll
  for (int i = 0; i < 8; ++i) { r[i] = lo[i]; r[i + 8] = hi[i]; }
  return r;
}

// B-fragment (32x16, K x N): lane l holds column (l&15); K chunks at (l>>4)*8 and (l>>4)*8+16
__device__ __forceinline__ v16bf fragB(const __bf16* sm, int lane) {
  int col = lane & 15;
  int kb  = (lane >> 4) << 3;
  const __bf16* p = sm + col * LS + kb;
  v8bf lo = *(const v8bf*)p;
  v8bf hi = *(const v8bf*)(p + 16);
  v16bf r;
#pragma unroll
  for (int i = 0; i < 8; ++i) { r[i] = lo[i]; r[i + 8] = hi[i]; }
  return r;
}

// ---------------- fp32 -> bf16 conversion ----------------

__global__ __launch_bounds__(256) void k_f2bf(const float* __restrict__ in,
                                              __bf16* __restrict__ out) {
  size_t i = ((size_t)blockIdx.x * 256 + threadIdx.x) * 4;
  float4 v = *(const float4*)(in + i);
  v4bf o;
  o[0] = f2bf(v.x); o[1] = f2bf(v.y); o[2] = f2bf(v.z); o[3] = f2bf(v.w);
  *(v4bf*)(out + i) = o;
}

// ---------------- big-tile GEMM: C = act(A @ W^T) ----------------
// Block tile 256(M) x 128(N), K-slice 32, 8 waves (4x2), wave tile 64x64.
// Async global->LDS double buffer, one barrier per K-step.
// MODE 0: silu epilogue -> bf16   MODE 1: plain -> fp32

template <int MODE>
__global__ __launch_bounds__(256) void k_gemm_big(const __bf16* __restrict__ A,
                                                  const __bf16* __restrict__ W,
                                                  void* __restrict__ Ov) {
  constexpr int BM = 256, BN = 128, BK = 32, NK = D_ / BK;
  __shared__ __bf16 sA[2][BM * LS];
  __shared__ __bf16 sB[2][BN * LS];
  const int tid = threadIdx.x, lane = tid & 31, wave = tid >> 5;
  const int wm = wave >> 1, wn = wave & 1;
  const int bm = blockIdx.y * BM, bn = blockIdx.x * BN;

  v8f acc[4][4];
#pragma unroll
  for (int mi = 0; mi < 4; ++mi)
#pragma unroll
    for (int ni = 0; ni < 4; ++ni) acc[mi][ni] = (v8f){0, 0, 0, 0, 0, 0, 0, 0};

  const int r4 = tid >> 2, c4 = (tid & 3) << 3;

#if ASYNC_LDS
  auto acopy = [&](int k0, int buf) {
#pragma unroll
    for (int it = 0; it < 4; ++it)
      ag2l128(A + (size_t)(bm + r4 + it * 64) * D_ + k0 + c4,
              &sA[buf][(r4 + it * 64) * LS + c4]);
#pragma unroll
    for (int it = 0; it < 2; ++it)
      ag2l128(W + (size_t)(bn + r4 + it * 64) * D_ + k0 + c4,
              &sB[buf][(r4 + it * 64) * LS + c4]);
  };
#else
  v8bf ra[4], rb[2];
  auto gload = [&](int k0) {
#pragma unroll
    for (int it = 0; it < 4; ++it)
      ra[it] = *(const v8bf*)(A + (size_t)(bm + r4 + it * 64) * D_ + k0 + c4);
#pragma unroll
    for (int it = 0; it < 2; ++it)
      rb[it] = *(const v8bf*)(W + (size_t)(bn + r4 + it * 64) * D_ + k0 + c4);
  };
  auto sstore = [&](int buf) {
#pragma unroll
    for (int it = 0; it < 4; ++it)
      *(v8bf*)(&sA[buf][(r4 + it * 64) * LS + c4]) = ra[it];
#pragma unroll
    for (int it = 0; it < 2; ++it)
      *(v8bf*)(&sB[buf][(r4 + it * 64) * LS + c4]) = rb[it];
  };
#endif

  auto compute = [&](int buf) {
    v16bf bfr[4];
#pragma unroll
    for (int ni = 0; ni < 4; ++ni) bfr[ni] = fragB(&sB[buf][(wn * 64 + ni * 16) * LS], lane);
#pragma unroll
    for (int mi = 0; mi < 4; ++mi) {
      v16bf afr = fragA(&sA[buf][(wm * 64 + mi * 16) * LS], lane);
#pragma unroll
      for (int ni = 0; ni < 4; ++ni)
        acc[mi][ni] = __builtin_amdgcn_wmma_f32_16x16x32_bf16(
            false, afr, false, bfr[ni], (short)0, acc[mi][ni], false, false);
    }
  };

#if ASYNC_LDS
  acopy(0, 0);
  wait_async();
  __syncthreads();
  for (int kt = 0; kt < NK; ++kt) {
    const bool more = (kt + 1 < NK);
    if (more) {
      acopy((kt + 1) * BK, (kt + 1) & 1);
      if (kt + 2 < NK) {
        __builtin_prefetch(A + (size_t)(bm + r4) * D_ + (kt + 2) * BK + c4, 0, 3);
        __builtin_prefetch(W + (size_t)(bn + r4) * D_ + (kt + 2) * BK + c4, 0, 3);
      }
    }
    compute(kt & 1);
    if (more) wait_async();
    __syncthreads();
  }
#else
  gload(0);
  sstore(0);
  __syncthreads();
  for (int kt = 0; kt < NK; ++kt) {
    const bool more = (kt + 1 < NK);
    if (more) {
      gload((kt + 1) * BK);
      if (kt + 2 < NK) {
        __builtin_prefetch(A + (size_t)(bm + r4) * D_ + (kt + 2) * BK + c4, 0, 3);
        __builtin_prefetch(W + (size_t)(bn + r4) * D_ + (kt + 2) * BK + c4, 0, 3);
      }
    }
    compute(kt & 1);
    if (more) sstore((kt + 1) & 1);
    __syncthreads();
  }
#endif

  const int ml = (lane >> 4) << 3, nl = lane & 15;
#pragma unroll
  for (int mi = 0; mi < 4; ++mi)
#pragma unroll
    for (int ni = 0; ni < 4; ++ni)
#pragma unroll
      for (int r = 0; r < 8; ++r) {
        int m = bm + wm * 64 + mi * 16 + ml + r;
        int n = bn + wn * 64 + ni * 16 + nl;
        float c = acc[mi][ni][r];
        if constexpr (MODE == 0) {
          ((__bf16*)Ov)[(size_t)m * D_ + n] = f2bf(c * sigm(c));
        } else {
          ((float*)Ov)[(size_t)m * D_ + n] = c;
        }
      }
}

// ---------------- fused gates GEMM: alpha, bv from xp ----------------
// Block tile 128x128, wave tile 64x32 with 3 accumulator sets (Wa/Wb/Wv share
// the A fragments). Async double buffer, one barrier per K-step.

__global__ __launch_bounds__(256) void k_gemm_gates(const __bf16* __restrict__ A,
                                                    const __bf16* __restrict__ Wa,
                                                    const __bf16* __restrict__ Wb,
                                                    const __bf16* __restrict__ Wv,
                                                    const float* __restrict__ biasA,
                                                    const float* __restrict__ biasB,
                                                    const float* __restrict__ biasV,
                                                    __bf16* __restrict__ Oal,
                                                    __bf16* __restrict__ Obv) {
  constexpr int BK = 32, NK = D_ / BK;
  __shared__ __bf16 sA[2][128 * LS];
  __shared__ __bf16 sBa[2][128 * LS];
  __shared__ __bf16 sBb[2][128 * LS];
  __shared__ __bf16 sBv[2][128 * LS];
  const int tid = threadIdx.x, lane = tid & 31, wave = tid >> 5;
  const int wm = wave >> 2, wn = wave & 3;
  const int bm = blockIdx.y * 128, bn = blockIdx.x * 128;

  v8f accA[4][2], accB[4][2], accV[4][2];
#pragma unroll
  for (int mi = 0; mi < 4; ++mi)
#pragma unroll
    for (int ni = 0; ni < 2; ++ni) {
      accA[mi][ni] = (v8f){0, 0, 0, 0, 0, 0, 0, 0};
      accB[mi][ni] = (v8f){0, 0, 0, 0, 0, 0, 0, 0};
      accV[mi][ni] = (v8f){0, 0, 0, 0, 0, 0, 0, 0};
    }

  const int r4 = tid >> 2, c4 = (tid & 3) << 3;

#if ASYNC_LDS
  auto acopy = [&](int k0, int buf) {
#pragma unroll
    for (int it = 0; it < 2; ++it) {
      size_t rowm = (size_t)(bm + r4 + it * 64) * D_ + k0 + c4;
      size_t rown = (size_t)(bn + r4 + it * 64) * D_ + k0 + c4;
      int o = (r4 + it * 64) * LS + c4;
      ag2l128(A  + rowm, &sA[buf][o]);
      ag2l128(Wa + rown, &sBa[buf][o]);
      ag2l128(Wb + rown, &sBb[buf][o]);
      ag2l128(Wv + rown, &sBv[buf][o]);
    }
  };
#else
  v8bf rA[2], rBa[2], rBb[2], rBv[2];
  auto gload = [&](int k0) {
#pragma unroll
    for (int it = 0; it < 2; ++it) {
      size_t rowm = (size_t)(bm + r4 + it * 64) * D_ + k0 + c4;
      size_t rown = (size_t)(bn + r4 + it * 64) * D_ + k0 + c4;
      rA[it]  = *(const v8bf*)(A + rowm);
      rBa[it] = *(const v8bf*)(Wa + rown);
      rBb[it] = *(const v8bf*)(Wb + rown);
      rBv[it] = *(const v8bf*)(Wv + rown);
    }
  };
  auto sstore = [&](int buf) {
#pragma unroll
    for (int it = 0; it < 2; ++it) {
      int o = (r4 + it * 64) * LS + c4;
      *(v8bf*)(&sA[buf][o])  = rA[it];
      *(v8bf*)(&sBa[buf][o]) = rBa[it];
      *(v8bf*)(&sBb[buf][o]) = rBb[it];
      *(v8bf*)(&sBv[buf][o]) = rBv[it];
    }
  };
#endif

  auto compute = [&](int buf) {
    v16bf ba[2], bb[2], bv[2];
#pragma unroll
    for (int ni = 0; ni < 2; ++ni) {
      int o = (wn * 32 + ni * 16) * LS;
      ba[ni] = fragB(&sBa[buf][o], lane);
      bb[ni] = fragB(&sBb[buf][o], lane);
      bv[ni] = fragB(&sBv[buf][o], lane);
    }
#pragma unroll
    for (int mi = 0; mi < 4; ++mi) {
      v16bf afr = fragA(&sA[buf][(wm * 64 + mi * 16) * LS], lane);
#pragma unroll
      for (int ni = 0; ni < 2; ++ni) {
        accA[mi][ni] = __builtin_amdgcn_wmma_f32_16x16x32_bf16(
            false, afr, false, ba[ni], (short)0, accA[mi][ni], false, false);
        accB[mi][ni] = __builtin_amdgcn_wmma_f32_16x16x32_bf16(
            false, afr, false, bb[ni], (short)0, accB[mi][ni], false, false);
        accV[mi][ni] = __builtin_amdgcn_wmma_f32_16x16x32_bf16(
            false, afr, false, bv[ni], (short)0, accV[mi][ni], false, false);
      }
    }
  };

#if ASYNC_LDS
  acopy(0, 0);
  wait_async();
  __syncthreads();
  for (int kt = 0; kt < NK; ++kt) {
    const bool more = (kt + 1 < NK);
    if (more) {
      acopy((kt + 1) * BK, (kt + 1) & 1);
      if (kt + 2 < NK)
        __builtin_prefetch(A + (size_t)(bm + r4) * D_ + (kt + 2) * BK + c4, 0, 3);
    }
    compute(kt & 1);
    if (more) wait_async();
    __syncthreads();
  }
#else
  gload(0);
  sstore(0);
  __syncthreads();
  for (int kt = 0; kt < NK; ++kt) {
    const bool more = (kt + 1 < NK);
    if (more) {
      gload((kt + 1) * BK);
      if (kt + 2 < NK)
        __builtin_prefetch(A + (size_t)(bm + r4) * D_ + (kt + 2) * BK + c4, 0, 3);
    }
    compute(kt & 1);
    if (more) sstore((kt + 1) & 1);
    __syncthreads();
  }
#endif

  const int ml = (lane >> 4) << 3, nl = lane & 15;
#pragma unroll
  for (int ni = 0; ni < 2; ++ni) {
    int n = bn + wn * 32 + ni * 16 + nl;
    float bA = biasA[n], bB = biasB[n], bV = biasV[n];
#pragma unroll
    for (int mi = 0; mi < 4; ++mi)
#pragma unroll
      for (int r = 0; r < 8; ++r) {
        int m = bm + wm * 64 + mi * 16 + ml + r;
        size_t o = (size_t)m * D_ + n;
        float alv = sigm(accA[mi][ni][r] + bA);
        float bev = sigm(accB[mi][ni][r] + bB);
        float vv  = tanh_fast(accV[mi][ni][r] + bV);
        Oal[o] = f2bf(alv);
        Obv[o] = f2bf(bev * vv);
      }
  }
}

// ---------------- sequential scan over T (parallel over B*D channels) ----------------

__global__ __launch_bounds__(256) void k_scan(const __bf16* __restrict__ al,
                                              const __bf16* __restrict__ bvw,
                                              __bf16* __restrict__ cell,
                                              float* __restrict__ hlast) {
  int idx = blockIdx.x * 256 + threadIdx.x;  // 0..2047, 2 channels each
  int b  = idx >> 9;                         // 512 pairs per batch
  int d2 = idx & 511;
  size_t base = (size_t)b * T_ * D_ + (size_t)d2 * 2;
  float h0 = 0.f, h1 = 0.f;
  for (int t = 0; t < T_; ++t) {
    size_t off = base + (size_t)t * D_;
    unsigned ua = *(const unsigned*)(al + off);
    unsigned uw = *(const unsigned*)(bvw + off);
    float a0 = bfu2f((unsigned short)ua), a1 = bfu2f((unsigned short)(ua >> 16));
    float w0 = bfu2f((unsigned short)uw), w1 = bfu2f((unsigned short)(uw >> 16));
    h0 = fmaf(a0, h0, w0);
    h1 = fmaf(a1, h1, w1);
    float o0 = h0 * h0 * sigm(h0);  // h * silu(h)
    float o1 = h1 * h1 * sigm(h1);
    unsigned short s0 = __builtin_bit_cast(unsigned short, f2bf(o0));
    unsigned short s1 = __builtin_bit_cast(unsigned short, f2bf(o1));
    *(unsigned*)(cell + off) = (unsigned)s0 | ((unsigned)s1 << 16);
  }
  hlast[(size_t)b * D_ + d2 * 2]     = h0;
  hlast[(size_t)b * D_ + d2 * 2 + 1] = h1;
}

// ---------------- host launch ----------------

extern "C" void kernel_launch(void* const* d_in, const int* in_sizes, int n_in,
                              void* d_out, int out_size, void* d_ws, size_t ws_size,
                              hipStream_t stream) {
  const float* x       = (const float*)d_in[0];
  const float* W_in    = (const float*)d_in[1];
  const float* W_alpha = (const float*)d_in[2];
  const float* b_alpha = (const float*)d_in[3];
  const float* W_beta  = (const float*)d_in[4];
  const float* b_beta  = (const float*)d_in[5];
  const float* W_v     = (const float*)d_in[6];
  const float* b_v     = (const float*)d_in[7];
  const float* W_out   = (const float*)d_in[8];

  float* out   = (float*)d_out;
  float* hlast = out + (size_t)BT_ * D_;

  size_t off = 0;
  char* ws = (char*)d_ws;
  auto alloc = [&](size_t elems) -> __bf16* {
    __bf16* p = (__bf16*)(ws + off);
    off += elems * sizeof(__bf16);
    return p;
  };
  __bf16* xb    = alloc((size_t)BT_ * D_);
  __bf16* wib   = alloc((size_t)D_ * D_);
  __bf16* wab   = alloc((size_t)D_ * D_);
  __bf16* wbb   = alloc((size_t)D_ * D_);
  __bf16* wvb   = alloc((size_t)D_ * D_);
  __bf16* wob   = alloc((size_t)D_ * D_);
  __bf16* xpb   = alloc((size_t)BT_ * D_);
  __bf16* alb   = alloc((size_t)BT_ * D_);
  __bf16* bvb   = alloc((size_t)BT_ * D_);
  __bf16* cellb = alloc((size_t)BT_ * D_);

  // fp32 -> bf16 staging (each size divisible by 1024 = 256 threads * 4 elems)
  k_f2bf<<<(BT_ * (size_t)D_) / 1024, 256, 0, stream>>>(x, xb);
  k_f2bf<<<((size_t)D_ * D_) / 1024, 256, 0, stream>>>(W_in, wib);
  k_f2bf<<<((size_t)D_ * D_) / 1024, 256, 0, stream>>>(W_alpha, wab);
  k_f2bf<<<((size_t)D_ * D_) / 1024, 256, 0, stream>>>(W_beta, wbb);
  k_f2bf<<<((size_t)D_ * D_) / 1024, 256, 0, stream>>>(W_v, wvb);
  k_f2bf<<<((size_t)D_ * D_) / 1024, 256, 0, stream>>>(W_out, wob);

  dim3 grid_big(D_ / 128, BT_ / 256);    // (8, 64)
  dim3 grid_gates(D_ / 128, BT_ / 128);  // (8, 128)
  k_gemm_big<0><<<grid_big, 256, 0, stream>>>(xb, wib, xpb);
  k_gemm_gates<<<grid_gates, 256, 0, stream>>>(xpb, wab, wbb, wvb, b_alpha, b_beta, b_v, alb, bvb);
  k_scan<<<(B_ * D_ / 2) / 256, 256, 0, stream>>>(alb, bvb, cellb, hlast);
  k_gemm_big<1><<<grid_big, 256, 0, stream>>>(cellb, wob, out);
}
